// Self_Position_Attention_90460601188889
// MI455X (gfx1250) — matrix operand order repaired and checked by reading, not compiled, not run on hardware
//
#include <hip/hip_runtime.h>
#include <hip/hip_fp16.h>

typedef __attribute__((ext_vector_type(16))) _Float16 v16h;
typedef __attribute__((ext_vector_type(8)))  _Float16 v8h;
typedef __attribute__((ext_vector_type(8)))  float    v8f;
typedef __attribute__((ext_vector_type(4)))  uint32_t su4;
typedef __attribute__((ext_vector_type(8)))  uint32_t su8;

#define C_DIM 128
#define HP    128
#define WP    128
#define N_TOK (HP * WP)      // 16384
#define C8    16
#define KD    32             // qk dim padded 16 -> 32 for 16x16x32 WMMA
#define BN    128            // j-chunk size
#define H_IN  256
#define W_IN  256

__device__ __forceinline__ v16h cat16(v8h lo, v8h hi) {
  return __builtin_shufflevector(lo, hi, 0,1,2,3,4,5,6,7,8,9,10,11,12,13,14,15);
}

// One TDM 2-D tile DMA global -> LDS (TENSORcnt-tracked, EXEC-ignored).
// D# per CDNA5 ISA: group0 = {count/lds/global/type},
// group1 = {data_size=2B, tensor dims, tile dims, dim0 stride}. 2-D -> 2 groups.
__device__ __forceinline__ void tdm_load_2d(uint32_t lds_addr, uint64_t gaddr,
                                            uint32_t tdim0, uint32_t tdim1,
                                            uint32_t tile0, uint32_t tile1,
                                            uint32_t stride0) {
  su4 g0;
  g0[0] = 1u;                                                   // count=1 (valid user D#)
  g0[1] = lds_addr;                                             // LDS byte address
  g0[2] = (uint32_t)gaddr;                                      // global_addr[31:0]
  g0[3] = ((uint32_t)(gaddr >> 32) & 0x01FFFFFFu) | (2u << 30); // addr[56:32] | type=2
  su8 g1;
  g1[0] = 0x00010000u;                                          // data_size=1 (2 bytes)
  g1[1] = (tdim0 & 0xFFFFu) << 16;                              // tensor_dim0[15:0]
  g1[2] = (tdim0 >> 16) | ((tdim1 & 0xFFFFu) << 16);            // dim0 hi | dim1 lo
  g1[3] = (tdim1 >> 16) | (tile0 << 16);                        // dim1 hi | tile_dim0
  g1[4] = tile1;                                                // tile_dim1 (tile_dim2=0)
  g1[5] = stride0;                                              // tensor_dim0_stride lo
  g1[6] = 0u;                                                   // stride0 hi | stride1 lo
  g1[7] = 0u;                                                   // stride1 hi
  asm volatile("tensor_load_to_lds %0, %1" :: "s"(g0), "s"(g1) : "memory");
}

// ---------------- K1: 2x2 average pool: x(1,128,256,256) -> yf[c][n], n = hp*128+wp
__global__ void k_avgpool(const float* __restrict__ x, float* __restrict__ yf) {
  int idx = blockIdx.x * blockDim.x + threadIdx.x;       // c*N + n
  if (idx >= C_DIM * N_TOK) return;
  int c  = idx >> 14;
  int n  = idx & (N_TOK - 1);
  int hp = n >> 7, wp = n & 127;
  const float* xp = x + ((size_t)c * H_IN + 2 * hp) * W_IN + 2 * wp;
  yf[idx] = 0.25f * (xp[0] + xp[1] + xp[W_IN] + xp[W_IN + 1]);
}

// ---------------- K2a: q,k projections -> f16, padded to KD, row-major [N][KD]
__global__ void k_proj_qk(const float* __restrict__ yf,
                          const float* __restrict__ Wq, const float* __restrict__ bq,
                          const float* __restrict__ Wk, const float* __restrict__ bk,
                          _Float16* __restrict__ qh, _Float16* __restrict__ kh) {
  __shared__ float sWq[C8 * C_DIM];
  __shared__ float sWk[C8 * C_DIM];
  for (int i = threadIdx.x; i < C8 * C_DIM; i += blockDim.x) { sWq[i] = Wq[i]; sWk[i] = Wk[i]; }
  __syncthreads();
  int n = blockIdx.x * blockDim.x + threadIdx.x;         // one token per thread
  float aq[C8], ak[C8];
#pragma unroll
  for (int o = 0; o < C8; ++o) { aq[o] = bq[o]; ak[o] = bk[o]; }
  for (int c = 0; c < C_DIM; ++c) {
    float v = yf[(size_t)c * N_TOK + n];
#pragma unroll
    for (int o = 0; o < C8; ++o) {
      aq[o] = fmaf(sWq[o * C_DIM + c], v, aq[o]);
      ak[o] = fmaf(sWk[o * C_DIM + c], v, ak[o]);
    }
  }
  _Float16* qrow = qh + (size_t)n * KD;
  _Float16* krow = kh + (size_t)n * KD;
#pragma unroll
  for (int o = 0; o < C8; ++o) { qrow[o] = (_Float16)aq[o]; krow[o] = (_Float16)ak[o]; }
#pragma unroll
  for (int o = C8; o < KD; ++o) { qrow[o] = (_Float16)0.f; krow[o] = (_Float16)0.f; }
}

// ---------------- K2b: v projection -> f16 transposed layout vth[c][n]
__global__ void k_proj_v(const float* __restrict__ yf, const float* __restrict__ Wv,
                         const float* __restrict__ bv, _Float16* __restrict__ vth) {
  int idx = blockIdx.x * blockDim.x + threadIdx.x;       // c*N + n
  if (idx >= C_DIM * N_TOK) return;
  int c = idx >> 14;
  int n = idx & (N_TOK - 1);
  const float* wr = Wv + (size_t)c * C_DIM;
  float acc = bv[c];
  for (int cc = 0; cc < C_DIM; ++cc)
    acc = fmaf(wr[cc], yf[(size_t)cc * N_TOK + n], acc);
  vth[idx] = (_Float16)acc;
}

// ---------------- K3: fused flash attention, TDM double-buffered staging.
// 128 WGs x 256 thr (8 waves x 16 rows each). Wave 0 drives the Tensor Data Mover.
__global__ __launch_bounds__(256, 1)
void k_attn(const _Float16* __restrict__ qh, const _Float16* __restrict__ kh,
            const _Float16* __restrict__ vth, float* __restrict__ obuf) {
  __shared__ _Float16 kt[2][BN * KD];        // 2 x  8 KB  [j][KD]
  __shared__ _Float16 vt[2][C_DIM * BN];     // 2 x 32 KB  [c][j]
  __shared__ _Float16 pbuf[8 * 16 * BN];     //     32 KB  per-wave [16][BN]

  const int tid  = threadIdx.x;
  const int wave = tid >> 5;
  const int lane = tid & 31;
  const int half = lane >> 4;               // 0/1 lane-half (K split of WMMA operands)
  const int l16  = lane & 15;
  const int m0   = blockIdx.x * 128 + wave * 16;   // first query row of this wave

  // two TDM descriptors per chunk: K (1 row x 4096 elem) + V (128 rows x 128 elem)
  auto issue_chunk = [&](int j0, int buf) {
    tdm_load_2d((uint32_t)(uintptr_t)(&kt[buf][0]),
                (uint64_t)(uintptr_t)kh + (uint64_t)j0 * (KD * 2),
                (uint32_t)(N_TOK * KD), 1u, (uint32_t)(BN * KD), 1u,
                (uint32_t)(N_TOK * KD));
    tdm_load_2d((uint32_t)(uintptr_t)(&vt[buf][0]),
                (uint64_t)(uintptr_t)vth + (uint64_t)j0 * 2,
                (uint32_t)N_TOK, (uint32_t)C_DIM, (uint32_t)BN, (uint32_t)C_DIM,
                (uint32_t)N_TOK);
  };

  // --- Q A-fragment (doc A-layout: row = l16; K chunks {half*8..+8, +16..+24})
  v16h aq;
  {
    const _Float16* qp = qh + (size_t)(m0 + l16) * KD + half * 8;
    v8h lo = *(const v8h*)qp;
    v8h hi = *(const v8h*)(qp + 16);
    aq = cat16(lo, hi);
  }

  const v8f zero8 = {0.f, 0.f, 0.f, 0.f, 0.f, 0.f, 0.f, 0.f};
  v8f O[8];
#pragma unroll
  for (int ct = 0; ct < 8; ++ct) O[ct] = zero8;
  float mrun[8], lrun[8];
#pragma unroll
  for (int g = 0; g < 8; ++g) { mrun[g] = -3.0e38f; lrun[g] = 0.f; }

  _Float16* pw = pbuf + wave * 16 * BN;
  const int nchunks = N_TOK / BN;           // 128

  if (wave == 0) issue_chunk(0, 0);         // prologue DMA

  for (int ci = 0; ci < nchunks; ++ci) {
    const int buf = ci & 1;
    if (wave == 0) {
      if (ci + 1 < nchunks) {
        issue_chunk((ci + 1) * BN, buf ^ 1);        // DMA next chunk into spare buffer
        __builtin_amdgcn_s_wait_tensorcnt(2);       // retire chunk ci's two TDM ops
      } else {
        __builtin_amdgcn_s_wait_tensorcnt(0);
      }
    }
    __syncthreads();                        // publish chunk-ci LDS to all waves

    const _Float16* ktb = &kt[buf][0];
    const _Float16* vtb = &vt[buf][0];

    // --- S = q @ k  (8 tiles of 16x16, K=32)
    v8f S[8];
#pragma unroll
    for (int jt = 0; jt < 8; ++jt) {
      // B-layout: col = l16; K chunks {half*8..+8, +16..+24}
      const _Float16* kp = ktb + (jt * 16 + l16) * KD + half * 8;
      v16h bk_ = cat16(*(const v8h*)kp, *(const v8h*)(kp + 16));
      S[jt] = __builtin_amdgcn_wmma_f32_16x16x32_f16(false, aq, false, bk_,
                                                     (short)0, zero8, false, false);
    }

    // --- online softmax (row r = g + half*8; reduce across 16-lane halves)
#pragma unroll
    for (int g = 0; g < 8; ++g) {
      float m = S[0][g];
#pragma unroll
      for (int jt = 1; jt < 8; ++jt) m = fmaxf(m, S[jt][g]);
      m = fmaxf(m, __shfl_xor(m, 1, 32));
      m = fmaxf(m, __shfl_xor(m, 2, 32));
      m = fmaxf(m, __shfl_xor(m, 4, 32));
      m = fmaxf(m, __shfl_xor(m, 8, 32));
      float mc = fmaxf(mrun[g], m);
      float sc = __expf(mrun[g] - mc);
      mrun[g]  = mc;
      lrun[g] *= sc;
#pragma unroll
      for (int ct = 0; ct < 8; ++ct) O[ct][g] *= sc;
      float rs = 0.f;
#pragma unroll
      for (int jt = 0; jt < 8; ++jt) {
        float p = __expf(S[jt][g] - mc);
        S[jt][g] = p;
        rs += p;
      }
      rs += __shfl_xor(rs, 1, 32);
      rs += __shfl_xor(rs, 2, 32);
      rs += __shfl_xor(rs, 4, 32);
      rs += __shfl_xor(rs, 8, 32);
      lrun[g] += rs;
      // spill P row (wave-private region, C/D layout -> [row][j])
      _Float16* pr = pw + (g + (half << 3)) * BN + l16;
#pragma unroll
      for (int jt = 0; jt < 8; ++jt) pr[jt * 16] = (_Float16)S[jt][g];
    }
    asm volatile("s_wait_dscnt 0" ::: "memory");   // wave-private LDS RAW

    // --- O += P @ V^T   (4 K-steps of 32 x 8 channel tiles)
#pragma unroll
    for (int kk = 0; kk < 4; ++kk) {
      const _Float16* pp = pw + l16 * BN + kk * 32 + half * 8;   // A-layout
      v16h ap = cat16(*(const v8h*)pp, *(const v8h*)(pp + 16));
#pragma unroll
      for (int ct = 0; ct < 8; ++ct) {
        const _Float16* vp = vtb + (ct * 16 + l16) * BN + kk * 32 + half * 8; // B-layout
        v16h bv_ = cat16(*(const v8h*)vp, *(const v8h*)(vp + 16));
        O[ct] = __builtin_amdgcn_wmma_f32_16x16x32_f16(false, ap, false, bv_,
                                                       (short)0, O[ct], false, false);
      }
    }
    __syncthreads();            // buffer `buf` free for TDM refill next iteration
  }

  // --- epilogue: normalize and store obuf[i][c]
#pragma unroll
  for (int g = 0; g < 8; ++g) {
    float inv = 1.0f / lrun[g];
    int r = m0 + g + (half << 3);
    float* orow = obuf + (size_t)r * C_DIM;
#pragma unroll
    for (int ct = 0; ct < 8; ++ct) orow[ct * 16 + l16] = O[ct][g] * inv;
  }
}

// ---------------- K4: bilinear x2 upsample (align_corners) + gamma*out + x
__global__ void k_upsample(const float* __restrict__ obuf, const float* __restrict__ x,
                           const float* __restrict__ gamma, float* __restrict__ out) {
  int idx = blockIdx.x * blockDim.x + threadIdx.x;
  if (idx >= C_DIM * H_IN * W_IN) return;
  int w = idx & 255;
  int h = (idx >> 8) & 255;
  int c = idx >> 16;
  const float rsc = 127.0f / 255.0f;
  float py = h * rsc; int y0 = (int)py; if (y0 > 126) y0 = 126; float wy = py - y0;
  float px = w * rsc; int x0 = (int)px; if (x0 > 126) x0 = 126; float wx = px - x0;
  const float* ob = obuf;                       // [i = y*WP + x][c]
  float v00 = ob[((size_t)(y0 * WP + x0)) * C_DIM + c];
  float v01 = ob[((size_t)(y0 * WP + x0 + 1)) * C_DIM + c];
  float v10 = ob[((size_t)((y0 + 1) * WP + x0)) * C_DIM + c];
  float v11 = ob[((size_t)((y0 + 1) * WP + x0 + 1)) * C_DIM + c];
  float top = v00 + (v01 - v00) * wx;
  float bot = v10 + (v11 - v10) * wx;
  float val = top + (bot - top) * wy;
  out[idx] = gamma[0] * val + x[idx];
}

extern "C" void kernel_launch(void* const* d_in, const int* in_sizes, int n_in,
                              void* d_out, int out_size, void* d_ws, size_t ws_size,
                              hipStream_t stream) {
  const float* x     = (const float*)d_in[0];
  const float* Wq    = (const float*)d_in[1];
  const float* bq    = (const float*)d_in[2];
  const float* Wk    = (const float*)d_in[3];
  const float* bk    = (const float*)d_in[4];
  const float* Wv    = (const float*)d_in[5];
  const float* bv    = (const float*)d_in[6];
  const float* gamma = (const float*)d_in[7];
  float* out = (float*)d_out;

  char* ws = (char*)d_ws;                         // 22 MB used
  float*    yf   = (float*)(ws);                  // [C][N]     8 MB
  float*    obuf = (float*)(ws + (8u  << 20));    // [N][C]     8 MB
  _Float16* qh   = (_Float16*)(ws + (16u << 20)); // [N][KD]    1 MB
  _Float16* kh   = (_Float16*)(ws + (17u << 20)); // [N][KD]    1 MB
  _Float16* vth  = (_Float16*)(ws + (18u << 20)); // [C][N]     4 MB

  k_avgpool <<<(C_DIM * N_TOK + 255) / 256, 256, 0, stream>>>(x, yf);
  k_proj_qk <<<N_TOK / 256,               256, 0, stream>>>(yf, Wq, bq, Wk, bk, qh, kh);
  k_proj_v  <<<(C_DIM * N_TOK + 255) / 256, 256, 0, stream>>>(yf, Wv, bv, vth);
  k_attn    <<<N_TOK / 128,               256, 0, stream>>>(qh, kh, vth, obuf);
  k_upsample<<<(C_DIM * H_IN * W_IN + 255) / 256, 256, 0, stream>>>(obuf, x, gamma, out);
}
